// MultiHeadAttentionNaive_71270687310592
// MI455X (gfx1250) — matrix operand order repaired and checked by reading, not compiled, not run on hardware
//
#include <hip/hip_runtime.h>

// Problem constants (match reference)
#define BB  4
#define SS  2048
#define EE  1024
#define HH  16
#define DHD 64

typedef __attribute__((ext_vector_type(16))) __bf16 v16bf;
typedef __attribute__((ext_vector_type(8)))  __bf16 v8bf;
typedef __attribute__((ext_vector_type(8)))  float  v8f;

// ---------------------------------------------------------------------------
// WMMA helper: D = A(16x32 bf16) * B(32x16 bf16) + C(16x16 f32)
// ---------------------------------------------------------------------------
static __device__ __forceinline__ v8f wmma_bf16(v16bf a, v16bf b, v8f c) {
  return __builtin_amdgcn_wmma_f32_16x16x32_bf16(
      /*neg_a=*/false, a, /*neg_b=*/false, b,
      /*c_mod=*/(short)0, c, /*reuse_a=*/false, /*reuse_b=*/false);
}

// A-fragment (16x32, 16-bit): lane L supplies row m = L&15.
// element e -> k = (e&7) + 8*(L>>4) + 16*(e>>3)  (ISA 7.12.2 table)
static __device__ __forceinline__ v16bf a_frag_from_f32(const float* row, int half) {
  const float* c0 = row + 8 * half;        // k = 8h + [0..7]
  const float* c1 = row + 16 + 8 * half;   // k = 16 + 8h + [0..7]
  v16bf a;
#pragma unroll
  for (int i = 0; i < 8; ++i) { a[i] = (__bf16)c0[i]; a[i + 8] = (__bf16)c1[i]; }
  return a;
}

static __device__ __forceinline__ v16bf a_frag_bf16(const __bf16* row, int half) {
  v8bf lo = *(const v8bf*)(row + 8 * half);
  v8bf hi = *(const v8bf*)(row + 16 + 8 * half);
  v16bf a;
#pragma unroll
  for (int i = 0; i < 8; ++i) { a[i] = lo[i]; a[i + 8] = hi[i]; }
  return a;
}

// B-fragment (32x16, 16-bit): lane L supplies column n = L&15.
// element e -> k = (e&7) + 8*(L>>4) + 16*(e>>3)  (same as A)
static __device__ __forceinline__ v16bf b_frag_from_f32(const float* col_k, int half) {
  const float* c0 = col_k + 8 * half;
  const float* c1 = col_k + 16 + 8 * half;
  v16bf b;
#pragma unroll
  for (int i = 0; i < 8; ++i) { b[i] = (__bf16)c0[i]; b[i + 8] = (__bf16)c1[i]; }
  return b;
}

static __device__ __forceinline__ v16bf b_frag_bf16(const __bf16* col_k, int half) {
  v8bf lo = *(const v8bf*)(col_k + 8 * half);
  v8bf hi = *(const v8bf*)(col_k + 16 + 8 * half);
  v16bf b;
#pragma unroll
  for (int i = 0; i < 8; ++i) { b[i] = lo[i]; b[i + 8] = hi[i]; }
  return b;
}

// ---------------------------------------------------------------------------
// Kernel 1: Q/K/V projection.  y[b,s,h,d] = sum_e x[b,s,e] * W[h,d,e]
// Register-blocked: one wave computes a 32(M) x 64(N=DH) block
// (2 A-frags x 4 B-frags -> 8 WMMAs per k-step).
// Q scaled by 1/sqrt(DH)=0.125 at store. V stored transposed [B,H,DH,S].
// ---------------------------------------------------------------------------
__global__ __launch_bounds__(32) void qkv_proj_kernel(
    const float* __restrict__ x,
    const float* __restrict__ Wq, const float* __restrict__ Wk,
    const float* __restrict__ Wv,
    __bf16* __restrict__ Q, __bf16* __restrict__ K, __bf16* __restrict__ Vt) {
  const int lane = threadIdx.x & 31;
  const int half = lane >> 4, lm = lane & 15;
  const int mt  = blockIdx.x;               // 32-row block of M = B*S
  const int sel = blockIdx.y / HH;          // 0=Q 1=K 2=V
  const int h   = blockIdx.y % HH;

  const float* W = (sel == 0 ? Wq : (sel == 1 ? Wk : Wv)) + (size_t)h * DHD * EE;

  const float* arow[2];
#pragma unroll
  for (int mi = 0; mi < 2; ++mi)
    arow[mi] = x + (size_t)(mt * 32 + mi * 16 + lm) * EE;
  const float* brow[4];
#pragma unroll
  for (int ni = 0; ni < 4; ++ni)
    brow[ni] = W + (size_t)(ni * 16 + lm) * EE;   // W[h][d][*] contiguous in e

  v8f c[2][4];
#pragma unroll
  for (int mi = 0; mi < 2; ++mi)
#pragma unroll
    for (int ni = 0; ni < 4; ++ni) c[mi][ni] = (v8f){};

  for (int k0 = 0; k0 < EE; k0 += 32) {
    __builtin_prefetch(arow[0] + k0 + 128, 0, 1);   // global_prefetch_b8
    v16bf a[2], b[4];
#pragma unroll
    for (int mi = 0; mi < 2; ++mi) a[mi] = a_frag_from_f32(arow[mi] + k0, half);
#pragma unroll
    for (int ni = 0; ni < 4; ++ni) b[ni] = b_frag_from_f32(brow[ni] + k0, half);
#pragma unroll
    for (int mi = 0; mi < 2; ++mi)
#pragma unroll
      for (int ni = 0; ni < 4; ++ni)
        c[mi][ni] = wmma_bf16(a[mi], b[ni], c[mi][ni]);
  }

  // C layout: lane holds (m = r + 8*half, n = lm) in c[..][..][r]
#pragma unroll
  for (int mi = 0; mi < 2; ++mi) {
#pragma unroll
    for (int r = 0; r < 8; ++r) {
      const int mg = mt * 32 + mi * 16 + r + 8 * half;
      const int b_ = mg / SS, s_ = mg % SS;
#pragma unroll
      for (int ni = 0; ni < 4; ++ni) {
        const int d = ni * 16 + lm;
        const float v = c[mi][ni][r];
        if (sel == 0) {
          Q[(((size_t)b_ * HH + h) * SS + s_) * DHD + d] = (__bf16)(v * 0.125f);
        } else if (sel == 1) {
          K[(((size_t)b_ * HH + h) * SS + s_) * DHD + d] = (__bf16)v;
        } else {
          Vt[(((size_t)b_ * HH + h) * DHD + d) * SS + s_] = (__bf16)v;
        }
      }
    }
  }
}

// ---------------------------------------------------------------------------
// Kernel 2: causal flash attention, one wave per (b,h, 32-query-row block).
// Two query m-tiles share every K/V fragment: 16 v_wmma per 32-key block.
// Online softmax; output written as bf16 head-concat [B,S,H,DH].
// ---------------------------------------------------------------------------
__global__ __launch_bounds__(32) void attn_kernel(
    const __bf16* __restrict__ Q, const __bf16* __restrict__ K,
    const __bf16* __restrict__ Vt, __bf16* __restrict__ attn) {
  __shared__ __bf16 Plds[32][40];            // 32x32 probs, padded rows
  const int lane = threadIdx.x & 31;
  const int half = lane >> 4, lm = lane & 15;
  const int qt = blockIdx.x;
  const int bh = blockIdx.y;                 // b*H + h
  const int b  = bh / HH, h = bh % HH;
  const int qs = qt * 32;

  const __bf16* Qb = Q  + ((size_t)bh * SS + qs) * DHD;
  const __bf16* Kb = K  + (size_t)bh * SS * DHD;
  const __bf16* Vb = Vt + (size_t)bh * DHD * SS;

  // Q A-fragments: [m-tile][d-step]  (row = lm within each 16-row tile)
  v16bf qa[2][2];
#pragma unroll
  for (int mi = 0; mi < 2; ++mi)
#pragma unroll
    for (int j = 0; j < 2; ++j)
      qa[mi][j] = a_frag_bf16(Qb + (size_t)(mi * 16 + lm) * DHD + j * 32, half);

  v8f o[2][4]; float Mrow[2][8], Lrow[2][8];
#pragma unroll
  for (int mi = 0; mi < 2; ++mi) {
#pragma unroll
    for (int dt = 0; dt < 4; ++dt) o[mi][dt] = (v8f){};
#pragma unroll
    for (int r = 0; r < 8; ++r) { Mrow[mi][r] = -3.0e38f; Lrow[mi][r] = 0.f; }
  }

  const float L2E = 1.44269504f;
  const int nkb = qt + 1;                    // key blocks covering keys <= qs+31

  for (int kb = 0; kb < nkb; ++kb) {
    const int ks = kb * 32;

    // K B-fragments: [t-tile][d-step], shared by both query m-tiles
    v16bf kf[2][2];
#pragma unroll
    for (int tt = 0; tt < 2; ++tt) {
      const __bf16* kp = Kb + (size_t)(ks + tt * 16 + lm) * DHD;
#pragma unroll
      for (int j = 0; j < 2; ++j) kf[tt][j] = b_frag_bf16(kp + j * 32, half);
    }

    // scores[mi][tt]
    v8f sc[2][2];
#pragma unroll
    for (int mi = 0; mi < 2; ++mi)
#pragma unroll
      for (int tt = 0; tt < 2; ++tt) {
        v8f s = (v8f){};
#pragma unroll
        for (int j = 0; j < 2; ++j) s = wmma_bf16(qa[mi][j], kf[tt][j], s);
        sc[mi][tt] = s;
      }

    // causal mask + online softmax per m-tile
#pragma unroll
    for (int mi = 0; mi < 2; ++mi) {
#pragma unroll
      for (int r = 0; r < 8; ++r) {
        const int mg = qs + mi * 16 + r + 8 * half;
        float s0 = (ks + lm      <= mg) ? sc[mi][0][r] : -1.0e30f;
        float s1 = (ks + 16 + lm <= mg) ? sc[mi][1][r] : -1.0e30f;
        float mx = fmaxf(s0, s1);
#pragma unroll
        for (int off = 1; off < 16; off <<= 1) mx = fmaxf(mx, __shfl_xor(mx, off));
        const float Mn = fmaxf(Mrow[mi][r], mx);
        const float al = __builtin_amdgcn_exp2f((Mrow[mi][r] - Mn) * L2E);
        const float p0 = __builtin_amdgcn_exp2f((s0 - Mn) * L2E);
        const float p1 = __builtin_amdgcn_exp2f((s1 - Mn) * L2E);
        float rs = p0 + p1;
#pragma unroll
        for (int off = 1; off < 16; off <<= 1) rs += __shfl_xor(rs, off);
        Lrow[mi][r] = Lrow[mi][r] * al + rs;
        Mrow[mi][r] = Mn;
#pragma unroll
        for (int dt = 0; dt < 4; ++dt) o[mi][dt][r] *= al;
        Plds[mi * 16 + r + 8 * half][lm]      = (__bf16)p0;
        Plds[mi * 16 + r + 8 * half][16 + lm] = (__bf16)p1;
      }
    }
    __syncthreads();

    // re-read probs as A-fragments (row = mi*16 + lm)
    v16bf pa[2];
#pragma unroll
    for (int mi = 0; mi < 2; ++mi) {
      const __bf16* prow = &Plds[mi * 16 + lm][0];
      v8bf lo = *(const v8bf*)(prow + 8 * half);
      v8bf hi = *(const v8bf*)(prow + 16 + 8 * half);
#pragma unroll
      for (int i = 0; i < 8; ++i) { pa[mi][i] = lo[i]; pa[mi][i + 8] = hi[i]; }
    }

    // V B-fragments loaded once, used by both m-tiles
#pragma unroll
    for (int dt = 0; dt < 4; ++dt) {
      v16bf vf = b_frag_bf16(Vb + (size_t)(dt * 16 + lm) * SS + ks, half);
#pragma unroll
      for (int mi = 0; mi < 2; ++mi)
        o[mi][dt] = wmma_bf16(pa[mi], vf, o[mi][dt]);
    }
    __syncthreads();
  }

  // normalize and store head-concat [B,S,H,DH]
#pragma unroll
  for (int mi = 0; mi < 2; ++mi) {
#pragma unroll
    for (int r = 0; r < 8; ++r) {
      const float inv = 1.0f / Lrow[mi][r];
      const int s_ = qs + mi * 16 + r + 8 * half;
#pragma unroll
      for (int dt = 0; dt < 4; ++dt) {
        const int d = dt * 16 + lm;
        attn[(((size_t)b * SS + s_) * HH + h) * DHD + d] =
            (__bf16)(o[mi][dt][r] * inv);
      }
    }
  }
}

// ---------------------------------------------------------------------------
// Kernel 3: output projection  out[m,n] = sum_k attn[m,k] * Wo[n,k]
// Register-blocked 32(M) x 64(N) per wave.
// ---------------------------------------------------------------------------
__global__ __launch_bounds__(32) void out_proj_kernel(
    const __bf16* __restrict__ attn, const float* __restrict__ Wo,
    float* __restrict__ out) {
  const int lane = threadIdx.x & 31;
  const int half = lane >> 4, lm = lane & 15;
  const int mt = blockIdx.x;                 // 32-row block
  const int nt = blockIdx.y;                 // 64-col block

  const __bf16* arow[2];
#pragma unroll
  for (int mi = 0; mi < 2; ++mi)
    arow[mi] = attn + (size_t)(mt * 32 + mi * 16 + lm) * EE;
  const float* brow[4];
#pragma unroll
  for (int ni = 0; ni < 4; ++ni)
    brow[ni] = Wo + (size_t)(nt * 64 + ni * 16 + lm) * EE;  // Wo[n][k]

  v8f c[2][4];
#pragma unroll
  for (int mi = 0; mi < 2; ++mi)
#pragma unroll
    for (int ni = 0; ni < 4; ++ni) c[mi][ni] = (v8f){};

  for (int k0 = 0; k0 < EE; k0 += 32) {
    __builtin_prefetch(arow[0] + k0 + 256, 0, 1);
    v16bf a[2], b[4];
#pragma unroll
    for (int mi = 0; mi < 2; ++mi) a[mi] = a_frag_bf16(arow[mi] + k0, half);
#pragma unroll
    for (int ni = 0; ni < 4; ++ni) b[ni] = b_frag_from_f32(brow[ni] + k0, half);
#pragma unroll
    for (int mi = 0; mi < 2; ++mi)
#pragma unroll
      for (int ni = 0; ni < 4; ++ni)
        c[mi][ni] = wmma_bf16(a[mi], b[ni], c[mi][ni]);
  }

#pragma unroll
  for (int mi = 0; mi < 2; ++mi)
#pragma unroll
    for (int r = 0; r < 8; ++r) {
      const size_t mg = (size_t)(mt * 32 + mi * 16 + r + 8 * half);
#pragma unroll
      for (int ni = 0; ni < 4; ++ni)
        out[mg * EE + nt * 64 + ni * 16 + lm] = c[mi][ni][r];
    }
}

// ---------------------------------------------------------------------------
// Host launcher.  Workspace layout (bf16): Q | K | V^T | attn-concat = 64 MB.
// ---------------------------------------------------------------------------
extern "C" void kernel_launch(void* const* d_in, const int* in_sizes, int n_in,
                              void* d_out, int out_size, void* d_ws, size_t ws_size,
                              hipStream_t stream) {
  const float* x  = (const float*)d_in[0];
  const float* Wq = (const float*)d_in[1];
  const float* Wk = (const float*)d_in[2];
  const float* Wv = (const float*)d_in[3];
  const float* Wo = (const float*)d_in[4];
  float* out = (float*)d_out;

  const size_t nElem = (size_t)BB * HH * SS * DHD;  // 8,388,608 per tensor
  __bf16* Q    = (__bf16*)d_ws;
  __bf16* K    = Q + nElem;
  __bf16* Vt   = K + nElem;
  __bf16* attn = Vt + nElem;

  dim3 g1(BB * SS / 32, 3 * HH);   // 256 x 48, one wave per block
  qkv_proj_kernel<<<g1, 32, 0, stream>>>(x, Wq, Wk, Wv, Q, K, Vt);

  dim3 g2(SS / 32, BB * HH);       // 64 x 64, one wave per block
  attn_kernel<<<g2, 32, 0, stream>>>(Q, K, Vt, attn);

  dim3 g3(BB * SS / 32, EE / 64);  // 256 x 16, one wave per block
  out_proj_kernel<<<g3, 32, 0, stream>>>(attn, Wo, out);

  (void)in_sizes; (void)n_in; (void)out_size; (void)ws_size;
}
